// NanoporeVQModel_42339787604295
// MI455X (gfx1250) — matrix operand order repaired and checked by reading, not compiled, not run on hardware
//
#include <hip/hip_runtime.h>

// ---------------------------------------------------------------------------
// Problem constants (from the reference)
// ---------------------------------------------------------------------------
#define BATCH   8
#define T_IN    24000
#define C1      16
#define C2      32
#define CZ      64
#define NZ      2000
#define M_ROWS  16000      // BATCH*NZ, rows of zf
#define NE      8192       // codebook size
#define T1      23989      // decoder length: 1999*12 - 24 + 25
#define M1      191912     // BATCH*T1
#define EPS_BN  1e-5f

// ---------------------------------------------------------------------------
// WMMA helpers (CDNA5, wave32)
// ---------------------------------------------------------------------------
typedef __attribute__((ext_vector_type(16))) _Float16 v16h;
typedef __attribute__((ext_vector_type(8)))  _Float16 v8h;
typedef __attribute__((ext_vector_type(8)))  float    v8f;

__device__ __forceinline__ v16h frag_cat(const _Float16* p0, const _Float16* p1) {
  v8h a = *(const v8h*)p0;
  v8h b = *(const v8h*)p1;
  v16h r;
#pragma unroll
  for (int i = 0; i < 8; ++i) { r[i] = a[i]; r[8 + i] = b[i]; }
  return r;
}
__device__ __forceinline__ v16h frag_zero() {
  v16h r;
#pragma unroll
  for (int i = 0; i < 16; ++i) r[i] = (_Float16)0.f;
  return r;
}
__device__ __forceinline__ v8f wmma_f16(v16h a, v16h b, v8f c) {
  // D(16x16,f32) = A(16x32,f16) x B(32x16,f16) + C
  return __builtin_amdgcn_wmma_f32_16x16x32_f16(false, a, false, b, (short)0, c,
                                                false, false);
}
__device__ __forceinline__ float silu(float x) { return x / (1.f + __expf(-x)); }
// raw v_sqrt_f32 (skip the IEEE range-fixup expansion; distances are O(10))
__device__ __forceinline__ float fast_sqrt(float x) { return __builtin_amdgcn_sqrtf(x); }

// Merge two online-softmax partials (min-dist d, rescaled sum S, argmin i)
// using a single exp.
__device__ __forceinline__ void vq_merge(float& d, float& S, int& i,
                                         float od, float oS, int oi) {
  float hi = fmaxf(d, od), lo = fminf(d, od);
  float f = __expf(-100.f * (hi - lo));
  bool oth = (od < d) || (od == d && oi < i);
  S = oth ? (oS + S * f) : (S + oS * f);
  if (oth) i = oi;
  d = lo;
}

// ---------------------------------------------------------------------------
// Encoder conv1: (8,1,24000) -> silu -> h1 (8,16,24000) channel-major
// ---------------------------------------------------------------------------
__global__ void k_conv1(const float* __restrict__ x, const float* __restrict__ w1,
                        const float* __restrict__ b1, float* __restrict__ h1) {
  __shared__ float sw[80], sb[16];
  int tid = threadIdx.x;
  if (tid < 80) sw[tid] = w1[tid];
  if (tid < 16) sb[tid] = b1[tid];
  __syncthreads();
  int i = blockIdx.x * 256 + tid;          // 192000 threads exact
  int b = i / T_IN, t = i % T_IN;
  float xv[5];
#pragma unroll
  for (int k = 0; k < 5; ++k) {
    int tt = t + k - 2;
    xv[k] = (tt >= 0 && tt < T_IN) ? x[(size_t)b * T_IN + tt] : 0.f;
  }
#pragma unroll
  for (int co = 0; co < 16; ++co) {
    float a = sb[co];
#pragma unroll
    for (int k = 0; k < 5; ++k) a += xv[k] * sw[co * 5 + k];
    h1[((size_t)(b * C1 + co)) * T_IN + t] = silu(a);
  }
}

// Per-channel mean/var over channel-major (B,C,T) f32 tensor. grid = C.
__global__ void k_stats_cm(const float* __restrict__ src, int C, int T, int B,
                           float* __restrict__ mv) {
  int c = blockIdx.x, tid = threadIdx.x;
  int n = B * T;
  float s = 0.f, s2 = 0.f;
  for (int i = tid; i < n; i += 256) {
    int b = i / T, t = i % T;
    float v = src[((size_t)(b * C + c)) * T + t];
    s += v; s2 += v * v;
  }
  __shared__ float ls[256], ls2[256];
  ls[tid] = s; ls2[tid] = s2; __syncthreads();
  for (int st = 128; st > 0; st >>= 1) {
    if (tid < st) { ls[tid] += ls[tid + st]; ls2[tid] += ls2[tid + st]; }
    __syncthreads();
  }
  if (tid == 0) { float m = ls[0] / n; mv[c] = m; mv[C + c] = ls2[0] / n - m * m; }
}

// Fold BN (stats mv over I input channels, gamma g, beta be) into next conv.
__global__ void k_prep_fold(const float* __restrict__ w, const float* __restrict__ bsrc,
                            const float* __restrict__ mv, const float* __restrict__ g,
                            const float* __restrict__ be, int O, int I, int K,
                            float* __restrict__ wdst, float* __restrict__ bdst) {
  __shared__ float sa[64], sc[64];
  int tid = threadIdx.x;
  if (tid < I) {
    float a = rsqrtf(mv[I + tid] + EPS_BN) * g[tid];
    sa[tid] = a;
    sc[tid] = be[tid] - mv[tid] * a;
  }
  __syncthreads();
  int n = O * I * K;
  for (int i = tid; i < n; i += blockDim.x) {
    int ci = (i / K) % I;
    wdst[i] = w[i] * sa[ci];
  }
  for (int co = tid; co < O; co += blockDim.x) {
    float bb = bsrc[co];
    for (int ci = 0; ci < I; ++ci) {
      const float* wp = w + ((size_t)co * I + ci) * K;
      float s = 0.f;
      for (int k = 0; k < K; ++k) s += wp[k];
      bb += s * sc[ci];
    }
    bdst[co] = bb;
  }
}

// conv2 (BN1 folded): h1(8,16,T) -> silu -> h2(8,32,T)
__global__ void k_conv2(const float* __restrict__ h1, const float* __restrict__ w2f,
                        const float* __restrict__ b2f, float* __restrict__ h2) {
  __shared__ float sw[32 * 16 * 5], sb[32];
  int tid = threadIdx.x;
  for (int i = tid; i < 2560; i += 256) sw[i] = w2f[i];
  if (tid < 32) sb[tid] = b2f[tid];
  __syncthreads();
  int i = blockIdx.x * 256 + tid;
  int b = i / T_IN, t = i % T_IN;
  float acc[32];
#pragma unroll
  for (int co = 0; co < 32; ++co) acc[co] = sb[co];
  for (int ci = 0; ci < 16; ++ci) {
    const float* hp = h1 + ((size_t)(b * C1 + ci)) * T_IN;
    float xv[5];
#pragma unroll
    for (int k = 0; k < 5; ++k) {
      int tt = t + k - 2;
      xv[k] = (tt >= 0 && tt < T_IN) ? hp[tt] : 0.f;
    }
#pragma unroll
    for (int co = 0; co < 32; ++co) {
      const float* wp = sw + (co * 16 + ci) * 5;
      acc[co] += xv[0]*wp[0] + xv[1]*wp[1] + xv[2]*wp[2] + xv[3]*wp[3] + xv[4]*wp[4];
    }
  }
#pragma unroll
  for (int co = 0; co < 32; ++co)
    h2[((size_t)(b * C2 + co)) * T_IN + t] = silu(acc[co]);
}

// conv3 (BN2 folded): stride 12, pad 12, k=25 -> tanh -> zraw (M_ROWS,64) row-major
__global__ void k_conv3(const float* __restrict__ h2, const float* __restrict__ w3f,
                        const float* __restrict__ b3f, float* __restrict__ zraw) {
  __shared__ float lx[32 * 205];
  int blk = blockIdx.x;                     // 8 * 125
  int b = blk / 125, nb = blk % 125;
  int tid = threadIdx.x;
  int t0 = nb * 16 * 12 - 12;
  for (int i = tid; i < 32 * 205; i += 256) {
    int ci = i / 205, s = i % 205;
    int tt = t0 + s;
    lx[i] = (tt >= 0 && tt < T_IN) ? h2[((size_t)(b * C2 + ci)) * T_IN + tt] : 0.f;
  }
  __syncthreads();
  int co = tid & 63, ng = tid >> 6;
  for (int j = 0; j < 4; ++j) {
    int nl = ng * 4 + j;
    int n = nb * 16 + nl;
    float acc = b3f[co];
    for (int ci = 0; ci < 32; ++ci) {
      const float* wp = w3f + ((size_t)co * 32 + ci) * 25;
      const float* xp = lx + ci * 205 + nl * 12;
      float s = 0.f;
#pragma unroll
      for (int k = 0; k < 25; ++k) s += xp[k] * wp[k];
      acc += s;
    }
    zraw[((size_t)(b * NZ + n)) * 64 + co] = tanhf(acc);
  }
}

// Two-stage per-channel stats for row-major (M,64) tensors (coalesced).
template <typename TT>
__global__ void k_colsum(const TT* __restrict__ src, int M, float* __restrict__ part) {
  int c = threadIdx.x;                 // 64 threads
  int nb = gridDim.x;
  int chunk = (M + nb - 1) / nb;
  int r0 = blockIdx.x * chunk;
  int r1 = min(M, r0 + chunk);
  float s = 0.f, s2 = 0.f;
  for (int r = r0; r < r1; ++r) {
    float v = (float)src[(size_t)r * 64 + c];
    s += v; s2 += v * v;
  }
  part[(size_t)blockIdx.x * 64 + c] = s;
  part[(size_t)nb * 64 + (size_t)blockIdx.x * 64 + c] = s2;
}
__global__ void k_colstat(const float* __restrict__ part, int nb, int M,
                          float* __restrict__ mv) {
  int c = threadIdx.x;                 // 64
  float s = 0.f, s2 = 0.f;
  for (int i = 0; i < nb; ++i) {
    s  += part[(size_t)i * 64 + c];
    s2 += part[(size_t)nb * 64 + (size_t)i * 64 + c];
  }
  float m = s / M;
  mv[c] = m; mv[64 + c] = s2 / M - m * m;
}

// Apply BN3 to zraw -> zf (f32) and zf_h (f16).
__global__ void k_bn3(const float* __restrict__ zraw, const float* __restrict__ mv,
                      const float* __restrict__ g, const float* __restrict__ be,
                      float* __restrict__ zf, _Float16* __restrict__ zfh) {
  int i = blockIdx.x * 256 + threadIdx.x;   // M_ROWS*64 exact
  int c = i & 63;
  float a = rsqrtf(mv[64 + c] + EPS_BN) * g[c];
  float v = (zraw[i] - mv[c]) * a + be[c];
  zf[i] = v;
  zfh[i] = (_Float16)v;
}

__global__ void k_zn2(const _Float16* __restrict__ zfh, float* __restrict__ zn2) {
  int r = blockIdx.x * 256 + threadIdx.x;
  if (r >= M_ROWS) return;
  const _Float16* p = zfh + (size_t)r * 64;
  float s = 0.f;
  for (int c = 0; c < 64; ++c) { float v = (float)p[c]; s += v * v; }
  zn2[r] = s;
}

__global__ void k_embed_prep(const float* __restrict__ embed, _Float16* __restrict__ eh,
                             float* __restrict__ en2) {
  int r = blockIdx.x * 256 + threadIdx.x;
  if (r >= NE) return;
  const float* p = embed + (size_t)r * 64;
  _Float16* q = eh + (size_t)r * 64;
  float s = 0.f;
  for (int c = 0; c < 64; ++c) {
    _Float16 h = (_Float16)p[c];
    q[c] = h;
    float v = (float)h;
    s += v * v;
  }
  en2[r] = s;
}

// ---------------------------------------------------------------------------
// VQ pass 1: WMMA distance GEMM + online (min-dist, argmin, softmax-Z) per row.
// grid 1000 (16 rows each), block 128 = 4 waves, each wave 128 column tiles.
// ---------------------------------------------------------------------------
__global__ void k_vq1(const _Float16* __restrict__ zfh, const _Float16* __restrict__ eh,
                      const float* __restrict__ zn2, const float* __restrict__ en2,
                      float* __restrict__ rowm, float* __restrict__ rowis,
                      int* __restrict__ rowidx) {
  int tid = threadIdx.x, lane = tid & 31, w = tid >> 5;
  int rbase = blockIdx.x * 16;
  int nl = lane & 15, half = lane >> 4;
  int arow = rbase + nl;
  int kb8 = half * 8;
  const _Float16* ar = zfh + (size_t)arow * 64;
  v16h a0 = frag_cat(ar + kb8,      ar + kb8 + 16);
  v16h a1 = frag_cat(ar + kb8 + 32, ar + kb8 + 48);
  float zn[8];
#pragma unroll
  for (int r = 0; r < 8; ++r) zn[r] = zn2[rbase + r + 8 * half];
  float bd[8], bS[8]; int bi[8];
#pragma unroll
  for (int r = 0; r < 8; ++r) { bd[r] = 3.0e38f; bS[r] = 0.f; bi[r] = 0; }
  int klo = half * 16;
  for (int j = 0; j < 128; ++j) {
    int tile = j * 4 + w;
    int n = tile * 16 + nl;
    const _Float16* bp = eh + (size_t)n * 64;
    if (j + 1 < 128)
      __builtin_prefetch(eh + ((size_t)((j + 1) * 4 + w) * 16 + nl) * 64, 0, 1);
    v16h b0 = frag_cat(bp + kb8,      bp + kb8 + 16);
    v16h b1 = frag_cat(bp + kb8 + 32, bp + kb8 + 48);
    v8f acc = {0.f, 0.f, 0.f, 0.f, 0.f, 0.f, 0.f, 0.f};
    acc = wmma_f16(a0, b0, acc);
    acc = wmma_f16(a1, b1, acc);
    float e2 = en2[n];
#pragma unroll
    for (int r = 0; r < 8; ++r) {
      float d2 = zn[r] + e2 - 2.f * acc[r];
      float d = fast_sqrt(fmaxf(d2, 1e-12f));
      float hi = fmaxf(bd[r], d), lo = fminf(bd[r], d);
      float f = __expf(-100.f * (hi - lo));          // single exp per element
      bool newmin = d < bd[r];
      bS[r] = newmin ? (1.f + bS[r] * f) : (bS[r] + f);
      if (newmin) bi[r] = n;
      bd[r] = lo;
    }
  }
  // butterfly over the 16 lanes of each half (same rows, different columns)
#pragma unroll
  for (int r = 0; r < 8; ++r) {
#pragma unroll
    for (int m = 1; m < 16; m <<= 1) {
      float od = __shfl_xor(bd[r], m, 32);
      float oS = __shfl_xor(bS[r], m, 32);
      int   oi = __shfl_xor(bi[r], m, 32);
      vq_merge(bd[r], bS[r], bi[r], od, oS, oi);
    }
  }
  __shared__ float sd[4][16], ss[4][16];
  __shared__ int   si[4][16];
  if (nl == 0) {
#pragma unroll
    for (int r = 0; r < 8; ++r) {
      int row = half * 8 + r;
      sd[w][row] = bd[r]; ss[w][row] = bS[r]; si[w][row] = bi[r];
    }
  }
  __syncthreads();
  if (w == 0 && lane < 16) {
    float d = sd[0][lane], S = ss[0][lane];
    int ii = si[0][lane];
    for (int q = 1; q < 4; ++q)
      vq_merge(d, S, ii, sd[q][lane], ss[q][lane], si[q][lane]);
    int row = rbase + lane;
    rowm[row] = d;
    rowis[row] = 1.f / S;
    rowidx[row] = ii;
  }
}

// VQ pass 2: recompute WMMA distances, accumulate per-column softmax partials
// (sum over this block's 16 rows) -> pprob[block][8192]. Deterministic.
__global__ void k_vq2(const _Float16* __restrict__ zfh, const _Float16* __restrict__ eh,
                      const float* __restrict__ zn2, const float* __restrict__ en2,
                      const float* __restrict__ rowm, const float* __restrict__ rowis,
                      float* __restrict__ pprob) {
  int tid = threadIdx.x, lane = tid & 31, w = tid >> 5;
  int rbase = blockIdx.x * 16;
  int nl = lane & 15, half = lane >> 4;
  int arow = rbase + nl;
  int kb8 = half * 8;
  const _Float16* ar = zfh + (size_t)arow * 64;
  v16h a0 = frag_cat(ar + kb8,      ar + kb8 + 16);
  v16h a1 = frag_cat(ar + kb8 + 32, ar + kb8 + 48);
  float zn[8], rm[8], ri[8];
#pragma unroll
  for (int r = 0; r < 8; ++r) {
    int row = rbase + r + 8 * half;
    zn[r] = zn2[row]; rm[r] = rowm[row]; ri[r] = rowis[row];
  }
  int klo = half * 16;
  for (int j = 0; j < 128; ++j) {
    int tile = j * 4 + w;
    int n = tile * 16 + nl;
    const _Float16* bp = eh + (size_t)n * 64;
    if (j + 1 < 128)
      __builtin_prefetch(eh + ((size_t)((j + 1) * 4 + w) * 16 + nl) * 64, 0, 1);
    v16h b0 = frag_cat(bp + kb8,      bp + kb8 + 16);
    v16h b1 = frag_cat(bp + kb8 + 32, bp + kb8 + 48);
    v8f acc = {0.f, 0.f, 0.f, 0.f, 0.f, 0.f, 0.f, 0.f};
    acc = wmma_f16(a0, b0, acc);
    acc = wmma_f16(a1, b1, acc);
    float e2 = en2[n];
    float csum = 0.f;
#pragma unroll
    for (int r = 0; r < 8; ++r) {
      float d2 = zn[r] + e2 - 2.f * acc[r];
      float d = fast_sqrt(fmaxf(d2, 1e-12f));
      csum += __expf(-100.f * (d - rm[r])) * ri[r];
    }
    csum += __shfl_xor(csum, 16, 32);      // merge the two row halves (same column)
    if (lane < 16)
      pprob[(size_t)blockIdx.x * NE + n] = csum;
  }
}

// Reduce pprob over the 1000 row blocks -> avg_prob; partial diversity sums.
__global__ void k_divred(const float* __restrict__ pprob, float* __restrict__ divpart) {
  int c = blockIdx.x * 256 + threadIdx.x;   // 8192 exact
  float s = 0.f;
  for (int g = 0; g < 1000; ++g) s += pprob[(size_t)g * NE + c];
  float avg = s / (float)M_ROWS;
  float contrib = avg * logf(fmaxf(avg, 1e-30f));
  __shared__ float ls[256];
  ls[threadIdx.x] = contrib; __syncthreads();
  for (int st = 128; st > 0; st >>= 1) {
    if (threadIdx.x < st) ls[threadIdx.x] += ls[threadIdx.x + st];
    __syncthreads();
  }
  if (threadIdx.x == 0) divpart[blockIdx.x] = ls[0];
}

__global__ void k_commit(const float* __restrict__ zf, const float* __restrict__ embed,
                         const int* __restrict__ rowidx, float* __restrict__ part) {
  int r = blockIdx.x * 256 + threadIdx.x;
  float s = 0.f;
  if (r < M_ROWS) {
    const float* zp = zf + (size_t)r * 64;
    const float* ep = embed + (size_t)rowidx[r] * 64;
    for (int c = 0; c < 64; ++c) { float d = ep[c] - zp[c]; s += d * d; }
  }
  __shared__ float ls[256];
  ls[threadIdx.x] = s; __syncthreads();
  for (int st = 128; st > 0; st >>= 1) {
    if (threadIdx.x < st) ls[threadIdx.x] += ls[threadIdx.x + st];
    __syncthreads();
  }
  if (threadIdx.x == 0) part[blockIdx.x] = ls[0];
}

__global__ void k_ortho(const float* __restrict__ embed, const int* __restrict__ oidx,
                        float* __restrict__ out) {
  __shared__ float rn[256];
  __shared__ int   ci[256];
  int tid = threadIdx.x;
  ci[tid] = oidx[tid];
  __syncthreads();
  {
    const float* p = embed + (size_t)ci[tid] * 64;
    float s = 0.f;
    for (int c = 0; c < 64; ++c) s += p[c] * p[c];
    rn[tid] = rsqrtf(s + 1e-12f);
  }
  __syncthreads();
  float acc = 0.f;
  for (int pq = tid; pq < 256 * 256; pq += 256) {
    int i = pq >> 8, j = pq & 255;
    const float* a = embed + (size_t)ci[i] * 64;
    const float* b = embed + (size_t)ci[j] * 64;
    float d = 0.f;
    for (int c = 0; c < 64; ++c) d += a[c] * b[c];
    float cs = d * rn[i] * rn[j];
    acc += cs * cs;
  }
  __shared__ float ls[256];
  ls[tid] = acc; __syncthreads();
  for (int st = 128; st > 0; st >>= 1) {
    if (tid < st) ls[tid] += ls[tid + st];
    __syncthreads();
  }
  if (tid == 0) out[0] = ls[0] / 65536.f - 1.f / 256.f;
}

// ---------------------------------------------------------------------------
// Decoder transposed conv as WMMA GEMM + tap-gather:
//   zq (16000,64 f16) gathered from codebook, then
//   G[m][k*64+co] = sum_ci zq[m][ci] * dw1[ci][co][k]  (M=16000, N=1600, K=64)
//   out[b][t][co]  = silu( sum_{k=(t+12)%12, k+=12, k<25} G[b*2000+n(t,k)][k*64+co] )
// ---------------------------------------------------------------------------
__global__ void k_zq(const _Float16* __restrict__ eh, const int* __restrict__ rowidx,
                     _Float16* __restrict__ zqh) {
  int r = blockIdx.x * 256 + threadIdx.x;
  if (r >= M_ROWS) return;
  const v8h* src = (const v8h*)(eh + (size_t)rowidx[r] * 64);
  v8h* dst = (v8h*)(zqh + (size_t)r * 64);
#pragma unroll
  for (int i = 0; i < 8; ++i) dst[i] = src[i];
}

__global__ void k_prep_dw1t(const float* __restrict__ dw1, _Float16* __restrict__ w1t) {
  int tid = threadIdx.x;                        // single block
  for (int i = tid; i < 25 * 64 * 64; i += 256) {
    int k = i / 4096, rem = i % 4096, co = rem >> 6, ci = rem & 63;
    w1t[i] = (_Float16)dw1[((size_t)ci * 64 + co) * 25 + k];   // -> [k][co][ci]
  }
}

// grid 1000 (16 rows), block 128 = 4 waves; wave w covers ntiles w,4+w,...
__global__ void k_dgemm(const _Float16* __restrict__ zqh, const _Float16* __restrict__ w1t,
                        _Float16* __restrict__ G) {
  int tid = threadIdx.x, lane = tid & 31, w = tid >> 5;
  int nl = lane & 15, half = lane >> 4;
  int rbase = blockIdx.x * 16;
  int kb8 = half * 8, klo = half * 16;
  const _Float16* ar = zqh + (size_t)(rbase + nl) * 64;
  v16h a0 = frag_cat(ar + kb8,      ar + kb8 + 16);
  v16h a1 = frag_cat(ar + kb8 + 32, ar + kb8 + 48);
  for (int j = 0; j < 25; ++j) {
    int n = (j * 4 + w) * 16 + nl;              // 0..1599
    const _Float16* bp = w1t + (size_t)n * 64;
    v16h b0 = frag_cat(bp + kb8,      bp + kb8 + 16);
    v16h b1 = frag_cat(bp + kb8 + 32, bp + kb8 + 48);
    v8f acc = {0.f, 0.f, 0.f, 0.f, 0.f, 0.f, 0.f, 0.f};
    acc = wmma_f16(a0, b0, acc);
    acc = wmma_f16(a1, b1, acc);
#pragma unroll
    for (int r = 0; r < 8; ++r) {
      int m = rbase + r + 8 * half;
      G[(size_t)m * 1600 + n] = (_Float16)acc[r];
    }
  }
}

// Sum the <=3 overlapping taps per output position, apply silu. Coalesced in co.
__global__ void k_dscat(const _Float16* __restrict__ G, _Float16* __restrict__ dec1h) {
  size_t i = (size_t)blockIdx.x * 256 + threadIdx.x;
  if (i >= (size_t)M1 * 64) return;
  int co = (int)(i & 63);
  size_t m = i >> 6;
  int t = (int)(m % T1), b = (int)(m / T1);
  float acc = 0.f;
  int r = (t + 12) % 12;
  for (int k = r; k < 25; k += 12) {
    int n = (t + 12 - k) / 12;
    if (n < 0 || n >= NZ) continue;
    acc += (float)G[(((size_t)(b * NZ + n)) * 25 + k) * 64 + co];
  }
  dec1h[i] = (_Float16)silu(acc);
}

// Fold decoder BN1 into dw2 and pre-transpose to w2t[k][co][ci] (f16).
__global__ void k_prep_dw2(const float* __restrict__ dw2, const float* __restrict__ db2,
                           const float* __restrict__ mvd1, const float* __restrict__ g,
                           const float* __restrict__ be, _Float16* __restrict__ w2t,
                           float* __restrict__ bias2) {
  __shared__ float sa[64], sc[64];
  int tid = threadIdx.x;
  if (tid < 64) {
    float a = rsqrtf(mvd1[64 + tid] + EPS_BN) * g[tid];
    sa[tid] = a;
    sc[tid] = be[tid] - mvd1[tid] * a;
  }
  __syncthreads();
  for (int i = tid; i < 5 * 64 * 64; i += 256) {
    int k = i / 4096, rem = i % 4096, co = rem >> 6, ci = rem & 63;
    w2t[i] = (_Float16)(dw2[((size_t)co * 64 + ci) * 5 + k] * sa[ci]);
  }
  for (int co = tid; co < 64; co += 256) {
    float bb = db2[co];
    for (int ci = 0; ci < 64; ++ci) {
      const float* wp = dw2 + ((size_t)co * 64 + ci) * 5;
      bb += (wp[0] + wp[1] + wp[2] + wp[3] + wp[4]) * sc[ci];
    }
    bias2[co] = bb;
  }
}

// Decoder 64->64 k=5 conv as 5 accumulated K=64 WMMA GEMMs.
// grid = Mtiles (11995), block 128 = 4 waves, wave w owns cout tile w.
__global__ void k_dw2(const _Float16* __restrict__ dec1h, const _Float16* __restrict__ w2t,
                      const float* __restrict__ bias2, _Float16* __restrict__ dec2h) {
  int tid = threadIdx.x, lane = tid & 31, w = tid >> 5;
  int mtile = blockIdx.x;
  int nl = lane & 15, half = lane >> 4;
  int co = w * 16 + nl;
  int mrow = mtile * 16 + nl;
  bool rowok = (mrow < M1);
  int b = 0, t = 0;
  if (rowok) { b = mrow / T1; t = mrow % T1; }
  int kb8 = half * 8;
  int klo = half * 16;
  v8f acc = {0.f, 0.f, 0.f, 0.f, 0.f, 0.f, 0.f, 0.f};
#pragma unroll
  for (int k = 0; k < 5; ++k) {
    int tt = t + k - 2;
    bool ok = rowok && tt >= 0 && tt < T1;
    const _Float16* ap = dec1h + ((size_t)b * T1 + (ok ? tt : 0)) * 64;
    const _Float16* bp = w2t + ((size_t)k * 64 + co) * 64;
    v16h a0 = ok ? frag_cat(ap + kb8,      ap + kb8 + 16) : frag_zero();
    v16h a1 = ok ? frag_cat(ap + kb8 + 32, ap + kb8 + 48) : frag_zero();
    v16h b0 = frag_cat(bp + kb8,      bp + kb8 + 16);
    v16h b1 = frag_cat(bp + kb8 + 32, bp + kb8 + 48);
    acc = wmma_f16(a0, b0, acc);
    acc = wmma_f16(a1, b1, acc);
  }
  float bb = bias2[co];
#pragma unroll
  for (int r = 0; r < 8; ++r) {
    int mo = mtile * 16 + r + 8 * half;
    if (mo < M1)
      dec2h[(size_t)mo * 64 + co] = (_Float16)silu(acc[r] + bb);
  }
}

// Fold decoder BN2 + 1x1 projection into alpha[c], beta.
__global__ void k_prep_recon(const float* __restrict__ mvd2, const float* __restrict__ g,
                             const float* __restrict__ be, const float* __restrict__ dw3,
                             const float* __restrict__ db3, float* __restrict__ alpha,
                             float* __restrict__ beta) {
  __shared__ float sct[64];
  int tid = threadIdx.x;
  if (tid < 64) {
    float a = rsqrtf(mvd2[64 + tid] + EPS_BN) * g[tid];
    alpha[tid] = a * dw3[tid];
    sct[tid] = (be[tid] - mvd2[tid] * a) * dw3[tid];
  }
  __syncthreads();
  if (tid == 0) {
    float s = db3[0];
    for (int c = 0; c < 64; ++c) s += sct[c];
    beta[0] = s;
  }
}

__global__ void k_recon(const _Float16* __restrict__ dec2h, const float* __restrict__ alpha,
                        const float* __restrict__ beta, float* __restrict__ out) {
  int i = blockIdx.x * 256 + threadIdx.x;   // BATCH*T_IN exact
  int b = i / T_IN, t = i % T_IN;
  float v = 0.f;
  if (t < T1) {
    const _Float16* p = dec2h + ((size_t)b * T1 + t) * 64;
    float s = beta[0];
    for (int c = 0; c < 64; ++c) s += (float)p[c] * alpha[c];
    v = s;
  }
  out[i] = v;
}

__global__ void k_idxout(const int* __restrict__ rowidx, float* __restrict__ out) {
  int r = blockIdx.x * 256 + threadIdx.x;
  if (r < M_ROWS) out[r] = (float)rowidx[r];
}

__global__ void k_finalize(const float* __restrict__ cpart, int ncp,
                           const float* __restrict__ dpart, int ndp,
                           const float* __restrict__ orthov, float* __restrict__ out) {
  if (threadIdx.x == 0) {
    float cm = 0.f;
    for (int i = 0; i < ncp; ++i) cm += cpart[i];
    cm /= (float)(M_ROWS * 64);
    float dv = 0.f;
    for (int i = 0; i < ndp; ++i) dv += dpart[i];
    float ot = orthov[0];
    out[0] = cm + dv + ot;   // loss
    out[1] = cm;
    out[2] = dv;
    out[3] = ot;
  }
}

// ---------------------------------------------------------------------------
// Host launch
// ---------------------------------------------------------------------------
extern "C" void kernel_launch(void* const* d_in, const int* in_sizes, int n_in,
                              void* d_out, int out_size, void* d_ws, size_t ws_size,
                              hipStream_t stream) {
  (void)in_sizes; (void)n_in; (void)out_size; (void)ws_size;
  const float* x     = (const float*)d_in[0];
  const float* w1    = (const float*)d_in[1];
  const float* b1    = (const float*)d_in[2];
  const float* g1    = (const float*)d_in[3];
  const float* be1   = (const float*)d_in[4];
  const float* w2    = (const float*)d_in[5];
  const float* b2    = (const float*)d_in[6];
  const float* g2    = (const float*)d_in[7];
  const float* be2   = (const float*)d_in[8];
  const float* w3    = (const float*)d_in[9];
  const float* b3    = (const float*)d_in[10];
  const float* g3    = (const float*)d_in[11];
  const float* be3   = (const float*)d_in[12];
  const float* embed = (const float*)d_in[13];
  const float* dw1   = (const float*)d_in[14];
  const float* dg1   = (const float*)d_in[15];
  const float* dbe1  = (const float*)d_in[16];
  const float* dw2   = (const float*)d_in[17];
  const float* db2   = (const float*)d_in[18];
  const float* dg2   = (const float*)d_in[19];
  const float* dbe2  = (const float*)d_in[20];
  const float* dw3   = (const float*)d_in[21];
  const float* db3   = (const float*)d_in[22];
  const int*   oidx  = (const int*)d_in[23];

  char* ws = (char*)d_ws;
  size_t off = 0;
  auto alloc = [&](size_t bytes) {
    size_t o = off;
    off = (off + bytes + 255) & ~(size_t)255;
    return o;
  };
  float*    h1     = (float*)(ws + alloc((size_t)BATCH * C1 * T_IN * 4));
  float*    h2     = (float*)(ws + alloc((size_t)BATCH * C2 * T_IN * 4));
  float*    zraw   = (float*)(ws + alloc((size_t)M_ROWS * 64 * 4));
  float*    zf     = (float*)(ws + alloc((size_t)M_ROWS * 64 * 4));
  _Float16* zfh    = (_Float16*)(ws + alloc((size_t)M_ROWS * 64 * 2));
  float*    zn2    = (float*)(ws + alloc((size_t)M_ROWS * 4));
  _Float16* eh     = (_Float16*)(ws + alloc((size_t)NE * 64 * 2));
  float*    en2    = (float*)(ws + alloc((size_t)NE * 4));
  float*    rowm   = (float*)(ws + alloc((size_t)M_ROWS * 4));
  float*    rowis  = (float*)(ws + alloc((size_t)M_ROWS * 4));
  int*      rowidx = (int*)(ws + alloc((size_t)M_ROWS * 4));
  float*    pprob  = (float*)(ws + alloc((size_t)1000 * NE * 4));
  _Float16* zqh    = (_Float16*)(ws + alloc((size_t)M_ROWS * 64 * 2));
  _Float16* w1t    = (_Float16*)(ws + alloc((size_t)25 * 64 * 64 * 2));
  _Float16* G      = (_Float16*)(ws + alloc((size_t)M_ROWS * 1600 * 2));
  _Float16* dec1h  = (_Float16*)(ws + alloc((size_t)M1 * 64 * 2));
  _Float16* dec2h  = (_Float16*)(ws + alloc((size_t)M1 * 64 * 2));
  float*    mv1    = (float*)(ws + alloc(2 * C1 * 4));
  float*    mv2    = (float*)(ws + alloc(2 * C2 * 4));
  float*    mv3    = (float*)(ws + alloc(2 * 64 * 4));
  float*    mvd1   = (float*)(ws + alloc(2 * 64 * 4));
  float*    mvd2   = (float*)(ws + alloc(2 * 64 * 4));
  float*    w2f    = (float*)(ws + alloc((size_t)32 * 16 * 5 * 4));
  float*    b2f    = (float*)(ws + alloc(32 * 4));
  float*    w3f    = (float*)(ws + alloc((size_t)64 * 32 * 25 * 4));
  float*    b3f    = (float*)(ws + alloc(64 * 4));
  _Float16* w2t    = (_Float16*)(ws + alloc((size_t)5 * 64 * 64 * 2));
  float*    bias2  = (float*)(ws + alloc(64 * 4));
  float*    alpha  = (float*)(ws + alloc(64 * 4));
  float*    beta   = (float*)(ws + alloc(4));
  float*    cpart  = (float*)(ws + alloc(64 * 4));
  float*    dpart  = (float*)(ws + alloc(32 * 4));
  float*    orthov = (float*)(ws + alloc(4));
  float*    cspart = (float*)(ws + alloc((size_t)2 * 128 * 64 * 4));

  float* out = (float*)d_out;
  float* out_recon = out;                  // 192000
  float* out_idx   = out + 192000;         // 16000
  float* out_loss  = out + 208000;         // 4

  // ---- encoder ----
  k_conv1<<<750, 256, 0, stream>>>(x, w1, b1, h1);
  k_stats_cm<<<C1, 256, 0, stream>>>(h1, C1, T_IN, BATCH, mv1);
  k_prep_fold<<<1, 256, 0, stream>>>(w2, b2, mv1, g1, be1, 32, 16, 5, w2f, b2f);
  k_conv2<<<750, 256, 0, stream>>>(h1, w2f, b2f, h2);
  k_stats_cm<<<C2, 256, 0, stream>>>(h2, C2, T_IN, BATCH, mv2);
  k_prep_fold<<<1, 256, 0, stream>>>(w3, b3, mv2, g2, be2, 64, 32, 25, w3f, b3f);
  k_conv3<<<1000, 256, 0, stream>>>(h2, w3f, b3f, zraw);
  k_colsum<float><<<128, 64, 0, stream>>>(zraw, M_ROWS, cspart);
  k_colstat<<<1, 64, 0, stream>>>(cspart, 128, M_ROWS, mv3);
  k_bn3<<<4000, 256, 0, stream>>>(zraw, mv3, g3, be3, zf, zfh);
  k_zn2<<<63, 256, 0, stream>>>(zfh, zn2);
  k_embed_prep<<<32, 256, 0, stream>>>(embed, eh, en2);

  // ---- VQ (WMMA distance GEMM x2) ----
  k_vq1<<<1000, 128, 0, stream>>>(zfh, eh, zn2, en2, rowm, rowis, rowidx);
  k_vq2<<<1000, 128, 0, stream>>>(zfh, eh, zn2, en2, rowm, rowis, pprob);
  k_divred<<<32, 256, 0, stream>>>(pprob, dpart);
  k_commit<<<63, 256, 0, stream>>>(zf, embed, rowidx, cpart);
  k_ortho<<<1, 256, 0, stream>>>(embed, oidx, orthov);

  // ---- decoder ----
  k_zq<<<63, 256, 0, stream>>>(eh, rowidx, zqh);
  k_prep_dw1t<<<1, 256, 0, stream>>>(dw1, w1t);
  k_dgemm<<<1000, 128, 0, stream>>>(zqh, w1t, G);
  k_dscat<<<47978, 256, 0, stream>>>(G, dec1h);
  k_colsum<_Float16><<<128, 64, 0, stream>>>(dec1h, M1, cspart);
  k_colstat<<<1, 64, 0, stream>>>(cspart, 128, M1, mvd1);
  k_prep_dw2<<<1, 256, 0, stream>>>(dw2, db2, mvd1, dg1, dbe1, w2t, bias2);
  k_dw2<<<11995, 128, 0, stream>>>(dec1h, w2t, bias2, dec2h);
  k_colsum<_Float16><<<128, 64, 0, stream>>>(dec2h, M1, cspart);
  k_colstat<<<1, 64, 0, stream>>>(cspart, 128, M1, mvd2);
  k_prep_recon<<<1, 64, 0, stream>>>(mvd2, dg2, dbe2, dw3, db3, alpha, beta);
  k_recon<<<750, 256, 0, stream>>>(dec2h, alpha, beta, out_recon);

  // ---- outputs ----
  k_idxout<<<63, 256, 0, stream>>>(rowidx, out_idx);
  k_finalize<<<1, 64, 0, stream>>>(cpart, 63, dpart, 32, orthov, out_loss);
}
